// SpatialGATEncoder_28174985461853
// MI455X (gfx1250) — matrix operand order repaired and checked by reading, compiled, not run on hardware
//
#include <hip/hip_runtime.h>
#include <hip/hip_bf16.h>
#include <stdint.h>

#define HID 512
#define NHEAD 8
#define DHEAD 64
#define NEG_SLOPE 0.2f

typedef __attribute__((ext_vector_type(16))) __bf16 v16bf;
typedef __attribute__((ext_vector_type(8)))  float  v8f;

union ABf { uint4 u[2]; v16bf v; };

#if defined(__has_builtin)
#if __has_builtin(__builtin_amdgcn_global_load_async_to_lds_b128)
#define HAVE_ASYNC_B128 1
#endif
#endif
#ifndef HAVE_ASYNC_B128
#define HAVE_ASYNC_B128 0
#endif

// pointer types matching the async-to-LDS builtin's signature:
// arg0: addrspace(1) pointer to 16B int vector; arg1: addrspace(3) same
typedef int v4i_ __attribute__((vector_size(16)));
typedef __attribute__((address_space(1))) v4i_* gptr_v4i;
typedef __attribute__((address_space(3))) v4i_* lptr_v4i;

__device__ __forceinline__ unsigned short f2bf(float f) {
  unsigned u = __float_as_uint(f);
  u += 0x7FFFu + ((u >> 16) & 1u);          // round-to-nearest-even
  return (unsigned short)(u >> 16);
}
// order-preserving float<->uint transform (atomic float max via atomicMax(u32))
__device__ __forceinline__ unsigned f2ord(float f) {
  unsigned u = __float_as_uint(f);
  return (u & 0x80000000u) ? ~u : (u | 0x80000000u);
}
__device__ __forceinline__ float ord2f(unsigned t) {
  unsigned b = (t & 0x80000000u) ? (t & 0x7FFFFFFFu) : ~t;
  return __uint_as_float(b);
}

// ---- gather embeddings + convert to bf16 (row-major [n, 512]) ----
__global__ __launch_bounds__(256)
void gather_bf16_k(const float* __restrict__ emb, const int* __restrict__ gid,
                   unsigned short* __restrict__ out, int n) {
  long long i = (long long)blockIdx.x * blockDim.x + threadIdx.x;
  if (i >= (long long)n * HID) return;
  int node = (int)(i >> 9);
  int c = (int)(i & (HID - 1));
  out[i] = f2bf(emb[(size_t)gid[node] * HID + c]);
}

// ---- pack W (512x512 f32) into per-lane WMMA B-fragment layout ----
// B frag (16x16x32 bf16): lane l holds column n = l&15; element e -> k = (l>>4)*16 + e
// storage: [(kb*32 + nt)*32 + lane]*16 contiguous bf16
__global__ __launch_bounds__(256)
void pack_w_k(const float* __restrict__ W, unsigned short* __restrict__ Bp) {
  int idx = blockIdx.x * blockDim.x + threadIdx.x;   // 16 kb * 32 nt * 32 lanes
  if (idx >= 16 * 32 * 32) return;
  int lane = idx & 31, nt = (idx >> 5) & 31, kb = idx >> 10;
  int n = nt * 16 + (lane & 15);
  int kbase = kb * 32 + (lane >> 4) * 16;
  unsigned short* dst = Bp + (size_t)idx * 16;
  #pragma unroll
  for (int e = 0; e < 16; ++e)
    dst[e] = f2bf(W[(size_t)(kbase + e) * HID + n]);
}

// ---- bf16 WMMA GEMM: C[M,512] = A[M,512] * W[512,512] ----
// Wave = 16(M) x 128(N); block = 8 waves = 128(M) x 128(N); grid.y = 4 over N.
// B panel (128 cols x 256 K = 64KB) staged in LDS per phase (2 phases over K)
// via async global->LDS b128. A frags load straight from global (two
// contiguous b128 per lane per K-step). B fragments double-buffered in
// registers so ds_load of frag t+1 overlaps the WMMA consuming frag t.
__global__ __launch_bounds__(256)
void gemm_bf16_k(const unsigned short* __restrict__ A, const unsigned short* __restrict__ Bp,
                 float* __restrict__ C, int M) {
  __shared__ uint4 ldsB[4096];                 // [kbl(8)][t(8)][lane(32)][half(2)] = 64KB
  const int tid  = threadIdx.x;
  const int lane = tid & 31;
  const int wave = tid >> 5;
  const int tileM = (blockIdx.x * 8 + wave) * 16;
  const int n0 = blockIdx.y * 128;
  const int khalf = lane >> 4;
  const bool waveActive = tileM < M;           // wave-uniform: EXEC stays all-1 for WMMA
  int rowA = tileM + (lane & 15);
  if (rowA >= M) rowA = 0;
  const unsigned short* Arow = A + (size_t)rowA * HID;
  const uint4* Bp4 = (const uint4*)Bp;

  v8f acc[8] = {};
  for (int p = 0; p < 2; ++p) {
    // cooperative fill: 4096 uint4 / 256 threads = 16 each, async global->LDS
    #pragma unroll
    for (int j = 0; j < 16; ++j) {
      int flat = j * 256 + tid;                // uint4 index in LDS
      int u = flat & 1;
      int frag = flat >> 1;                    // (kbl*8 + t)*32 + lane_c
      int lane_c = frag & 31;
      int t = (frag >> 5) & 7;
      int kbl = frag >> 8;
      size_t gidx = (((size_t)(p * 8 + kbl) * 32 + (n0 >> 4) + t) * 32 + lane_c) * 2 + u;
#if HAVE_ASYNC_B128
      __builtin_amdgcn_global_load_async_to_lds_b128(
          (gptr_v4i)(uint4*)&Bp4[gidx], (lptr_v4i)&ldsB[flat], 0, 0);
#else
      ldsB[flat] = Bp4[gidx];
#endif
    }
#if HAVE_ASYNC_B128
    asm volatile("s_wait_asynccnt 0" ::: "memory");
#endif
    __syncthreads();

    if (waveActive) {
      ABf bcur, bnxt;
      {
        const int base0 = lane * 2;            // kbl=0, t=0
        bcur.u[0] = ldsB[base0];
        bcur.u[1] = ldsB[base0 + 1];
      }
      for (int kbl = 0; kbl < 8; ++kbl) {
        const int k0 = (p * 8 + kbl) * 32;
        ABf a;
        a.u[0] = *(const uint4*)(Arow + k0 + khalf * 16);
        a.u[1] = *(const uint4*)(Arow + k0 + khalf * 16 + 8);
        #pragma unroll
        for (int t = 0; t < 8; ++t) {
          // prefetch next fragment (wraps at phase end; extra read is discarded)
          const int nk = (t == 7) ? ((kbl + 1) & 7) : kbl;
          const int nt = (t + 1) & 7;
          const int nbase = ((nk * 8 + nt) * 32 + lane) * 2;
          bnxt.u[0] = ldsB[nbase];
          bnxt.u[1] = ldsB[nbase + 1];
          acc[t] = __builtin_amdgcn_wmma_f32_16x16x32_bf16(
              false, a.v, false, bcur.v, (short)0, acc[t], false, false);
          bcur = bnxt;
        }
      }
    }
    __syncthreads();
  }
  if (!waveActive) return;

  // D layout: vgpr v, lane l -> m = (l>>4)*8 + v, n = l&15
  const int mBase = tileM + (lane >> 4) * 8;
  #pragma unroll
  for (int t = 0; t < 8; ++t) {
    const int col = n0 + t * 16 + (lane & 15);
    #pragma unroll
    for (int v = 0; v < 8; ++v) {
      const int m = mBase + v;
      if (m < M) C[(size_t)m * HID + col] = acc[t][v];
    }
  }
}

// ---- el/er: out[n,h] = sum_d f[n, h*64+d] * a[h*64+d] ----
__global__ __launch_bounds__(256)
void attn_dot_k(const float* __restrict__ f, const float* __restrict__ a,
                float* __restrict__ out, int n) {
  int i = blockIdx.x * blockDim.x + threadIdx.x;
  if (i >= n * NHEAD) return;
  int h = i & (NHEAD - 1);
  int node = i >> 3;
  const float* fp = f + (size_t)node * HID + h * DHEAD;
  const float* ap = a + h * DHEAD;
  float s = 0.f;
  #pragma unroll 8
  for (int d = 0; d < DHEAD; ++d) s += fp[d] * ap[d];
  out[i] = s;
}

// ---- segment max over dst (atomic u32 max on order-preserving encoding) ----
__global__ __launch_bounds__(256)
void edge_max_k(const int* __restrict__ src, const int* __restrict__ dst,
                const float* __restrict__ el, const float* __restrict__ er,
                unsigned* __restrict__ emax, int E) {
  int i = blockIdx.x * blockDim.x + threadIdx.x;
  if (i >= E * NHEAD) return;
  int h = i & (NHEAD - 1);
  int e = i >> 3;
  int s = src[e], d = dst[e];
  float x = el[s * NHEAD + h] + er[d * NHEAD + h];
  x = x < 0.f ? NEG_SLOPE * x : x;
  atomicMax(&emax[d * NHEAD + h], f2ord(x));
}

// ---- segment sum of exp(e - emax) ----
__global__ __launch_bounds__(256)
void edge_denom_k(const int* __restrict__ src, const int* __restrict__ dst,
                  const float* __restrict__ el, const float* __restrict__ er,
                  const unsigned* __restrict__ emax, float* __restrict__ den, int E) {
  int i = blockIdx.x * blockDim.x + threadIdx.x;
  if (i >= E * NHEAD) return;
  int h = i & (NHEAD - 1);
  int e = i >> 3;
  int s = src[e], d = dst[e];
  float x = el[s * NHEAD + h] + er[d * NHEAD + h];
  x = x < 0.f ? NEG_SLOPE * x : x;
  atomicAdd(&den[d * NHEAD + h], __expf(x - ord2f(emax[d * NHEAD + h])));
}

// ---- message aggregation: out[dst] += fs[src] * alpha (float4 per thread) ----
__global__ __launch_bounds__(256)
void edge_agg_k(const int* __restrict__ src, const int* __restrict__ dst,
                const float* __restrict__ el, const float* __restrict__ er,
                const unsigned* __restrict__ emax, const float* __restrict__ den,
                const float* __restrict__ fsrc, float* __restrict__ out,
                int E, int ostride) {
  long long i = (long long)blockIdx.x * blockDim.x + threadIdx.x;
  if (i >= (long long)E * (HID / 4)) return;
  int c = (int)(i & (HID / 4 - 1));   // 0..127 -> 4 floats each
  int e = (int)(i >> 7);
  int h = c >> 4;                     // (c*4)/64
  int s = src[e], d = dst[e];
  float x = el[s * NHEAD + h] + er[d * NHEAD + h];
  x = x < 0.f ? NEG_SLOPE * x : x;
  float alpha = __expf(x - ord2f(emax[d * NHEAD + h])) / den[d * NHEAD + h];
  const float4 f = *(const float4*)(fsrc + (size_t)s * HID + c * 4);
  float* o = out + (size_t)d * ostride + c * 4;
  atomicAdd(o + 0, f.x * alpha);
  atomicAdd(o + 1, f.y * alpha);
  atomicAdd(o + 2, f.z * alpha);
  atomicAdd(o + 3, f.w * alpha);
}

// ---- column sums of out_item [rows, 512] ----
__global__ __launch_bounds__(256)
void col_sum_k(const float* __restrict__ x, float* __restrict__ sum, int rows) {
  int col = (blockIdx.x & 1) * 256 + threadIdx.x;   // grid.x = 64 -> 32 row chunks
  int chunk = blockIdx.x >> 1;
  int per = (rows + 31) / 32;
  int r0 = chunk * per;
  int r1 = r0 + per; if (r1 > rows) r1 = rows;
  float s = 0.f;
  for (int r = r0; r < r1; ++r) s += x[(size_t)r * HID + col];
  atomicAdd(&sum[col], s);
}

// ---- finalize: user bias add + broadcast item mean (+item bias) ----
__global__ __launch_bounds__(256)
void finalize_k(float* __restrict__ out, const float* __restrict__ b_iu,
                const float* __restrict__ colsum, const float* __restrict__ b_ui,
                int U, float invI) {
  long long i = (long long)blockIdx.x * blockDim.x + threadIdx.x;
  if (i >= (long long)U * HID) return;
  int j = (int)(i & (HID - 1));
  long long u = i >> 9;
  float* row = out + u * (2 * HID);
  row[j] += b_iu[j];
  row[HID + j] = colsum[j] * invI + b_ui[j];
}

extern "C" void kernel_launch(void* const* d_in, const int* in_sizes, int n_in,
                              void* d_out, int out_size, void* d_ws, size_t ws_size,
                              hipStream_t stream) {
  const int*   u_gid    = (const int*)d_in[0];
  const int*   i_gid    = (const int*)d_in[1];
  const int*   src_u    = (const int*)d_in[2];
  const int*   dst_i    = (const int*)d_in[3];
  const float* user_emb = (const float*)d_in[4];
  const float* item_emb = (const float*)d_in[5];
  const float* W_ui     = (const float*)d_in[6];
  const float* al_ui    = (const float*)d_in[7];
  const float* ar_ui    = (const float*)d_in[8];
  const float* b_ui     = (const float*)d_in[9];
  const float* W_iu     = (const float*)d_in[10];
  const float* al_iu    = (const float*)d_in[11];
  const float* ar_iu    = (const float*)d_in[12];
  const float* b_iu     = (const float*)d_in[13];

  const int U = in_sizes[0];
  const int I = in_sizes[1];
  const int E = in_sizes[2];

  char* w = (char*)d_ws;
  auto carve = [&](size_t bytes) -> char* {
    char* p = w; w += (bytes + 255) & ~(size_t)255; return p;
  };
  unsigned short* hu_bf  = (unsigned short*)carve((size_t)U * HID * 2);
  unsigned short* hi_bf  = (unsigned short*)carve((size_t)I * HID * 2);
  unsigned short* Wui_p  = (unsigned short*)carve((size_t)HID * HID * 2);
  unsigned short* Wiu_p  = (unsigned short*)carve((size_t)HID * HID * 2);
  float*    fs_ui  = (float*)carve((size_t)U * HID * 4);
  float*    fd_ui  = (float*)carve((size_t)I * HID * 4);
  float*    fs_iu  = (float*)carve((size_t)I * HID * 4);
  float*    fd_iu  = (float*)carve((size_t)U * HID * 4);
  float*    el_ui  = (float*)carve((size_t)U * NHEAD * 4);
  float*    er_ui  = (float*)carve((size_t)I * NHEAD * 4);
  float*    el_iu  = (float*)carve((size_t)I * NHEAD * 4);
  float*    er_iu  = (float*)carve((size_t)U * NHEAD * 4);
  unsigned* emax_i = (unsigned*)carve((size_t)I * NHEAD * 4);
  unsigned* emax_u = (unsigned*)carve((size_t)U * NHEAD * 4);
  float*    den_i  = (float*)carve((size_t)I * NHEAD * 4);
  float*    den_u  = (float*)carve((size_t)U * NHEAD * 4);
  float*    out_it = (float*)carve((size_t)I * HID * 4);
  float*    colsum = (float*)carve(HID * 4);
  (void)ws_size; (void)n_in;

  (void)hipMemsetAsync(d_out, 0, (size_t)out_size * 4, stream);
  (void)hipMemsetAsync(out_it, 0, (size_t)I * HID * 4, stream);
  (void)hipMemsetAsync(den_i, 0, (size_t)I * NHEAD * 4, stream);
  (void)hipMemsetAsync(den_u, 0, (size_t)U * NHEAD * 4, stream);
  (void)hipMemsetAsync(emax_i, 0, (size_t)I * NHEAD * 4, stream);
  (void)hipMemsetAsync(emax_u, 0, (size_t)U * NHEAD * 4, stream);
  (void)hipMemsetAsync(colsum, 0, HID * 4, stream);

  long long tU = (long long)U * HID, tI = (long long)I * HID;
  gather_bf16_k<<<(unsigned)((tU + 255) / 256), 256, 0, stream>>>(user_emb, u_gid, hu_bf, U);
  gather_bf16_k<<<(unsigned)((tI + 255) / 256), 256, 0, stream>>>(item_emb, i_gid, hi_bf, I);
  pack_w_k<<<64, 256, 0, stream>>>(W_ui, Wui_p);
  pack_w_k<<<64, 256, 0, stream>>>(W_iu, Wiu_p);

  dim3 gU((U + 127) / 128, 4), gI((I + 127) / 128, 4);
  gemm_bf16_k<<<gU, 256, 0, stream>>>(hu_bf, Wui_p, fs_ui, U);
  gemm_bf16_k<<<gI, 256, 0, stream>>>(hi_bf, Wui_p, fd_ui, I);
  gemm_bf16_k<<<gI, 256, 0, stream>>>(hi_bf, Wiu_p, fs_iu, I);
  gemm_bf16_k<<<gU, 256, 0, stream>>>(hu_bf, Wiu_p, fd_iu, U);

  attn_dot_k<<<(U * NHEAD + 255) / 256, 256, 0, stream>>>(fs_ui, al_ui, el_ui, U);
  attn_dot_k<<<(I * NHEAD + 255) / 256, 256, 0, stream>>>(fd_ui, ar_ui, er_ui, I);
  attn_dot_k<<<(I * NHEAD + 255) / 256, 256, 0, stream>>>(fs_iu, al_iu, el_iu, I);
  attn_dot_k<<<(U * NHEAD + 255) / 256, 256, 0, stream>>>(fd_iu, ar_iu, er_iu, U);

  int ehB = (E * NHEAD + 255) / 256;
  long long aggT = (long long)E * (HID / 4);
  unsigned aggB = (unsigned)((aggT + 255) / 256);

  // layer 1: user -> item (dst = items)
  edge_max_k<<<ehB, 256, 0, stream>>>(src_u, dst_i, el_ui, er_ui, emax_i, E);
  edge_denom_k<<<ehB, 256, 0, stream>>>(src_u, dst_i, el_ui, er_ui, emax_i, den_i, E);
  edge_agg_k<<<aggB, 256, 0, stream>>>(src_u, dst_i, el_ui, er_ui, emax_i, den_i,
                                       fs_ui, out_it, E, HID);
  // layer 2: item -> user (reversed edges; dst = users) -> scatter into d_out stride 1024
  edge_max_k<<<ehB, 256, 0, stream>>>(dst_i, src_u, el_iu, er_iu, emax_u, E);
  edge_denom_k<<<ehB, 256, 0, stream>>>(dst_i, src_u, el_iu, er_iu, emax_u, den_u, E);
  edge_agg_k<<<aggB, 256, 0, stream>>>(dst_i, src_u, el_iu, er_iu, emax_u, den_u,
                                       fs_iu, (float*)d_out, E, 2 * HID);

  col_sum_k<<<64, 256, 0, stream>>>(out_it, colsum, I);
  finalize_k<<<(unsigned)((tU + 255) / 256), 256, 0, stream>>>(
      (float*)d_out, b_iu, colsum, b_ui, U, 1.0f / (float)I);
}
